// AttnPredictor_80582176407597
// MI455X (gfx1250) — matrix operand order repaired and checked by reading, not compiled, not run on hardware
//
#include <hip/hip_runtime.h>

// ---------------------------------------------------------------------------
// Problem constants (from reference)
// ---------------------------------------------------------------------------
static constexpr int NB   = 512;   // batch
static constexpr int ND   = 32;    // latent dims == heads
static constexpr int NA   = 18;    // actions
static constexpr int FED  = 128;
static constexpr int AED  = 64;
static constexpr int VD   = FED + AED;  // 192
static constexpr int KD   = VD;         // 192
static constexpr int HID  = 256;
static constexpr int NH   = ND;         // 32 heads

typedef __attribute__((ext_vector_type(16))) _Float16 v16h;
typedef __attribute__((ext_vector_type(8)))  float    v8f;

// ---------------------------------------------------------------------------
// WMMA helpers (wave32, v_wmma_f32_16x16x32_f16)
//   A layout (16-bit A 16x32, matched to B):  m = lane&15 ; lanes 0-15 hold
//   K 0..15, lanes 16-31 hold K 16..31.
//   B layout (32x16): n = lane&15, lanes 0-15 hold K 0..15, lanes 16-31 hold
//   K 16..31 (2 K per VGPR) -> with B stored pre-transposed [N][K] each lane
//   reads 2x16 contiguous bytes (global_load_b128).
//   C/D layout: n = lane&15, VGPR r -> m = r + 8*(lane>>4).
// ---------------------------------------------------------------------------
__device__ __forceinline__ v8f wmma_f16(v16h a, v16h b, v8f c) {
  return __builtin_amdgcn_wmma_f32_16x16x32_f16(
      /*neg_a=*/false, a, /*neg_b=*/false, b,
      /*c_mod=*/(short)0, c, /*reuse_a=*/false, /*reuse_b=*/false);
}

// Load 16x32 A tile from row-major f16. `tile` -> (0,0), ld in elements.
__device__ __forceinline__ v16h ldA_f16(const _Float16* tile, int ld) {
  int lane = threadIdx.x & 31;
  const _Float16* row = tile + (lane & 15) * ld + ((lane >> 4) << 4);
  v16h a;
#pragma unroll
  for (int i = 0; i < 16; ++i) {
    a[i]     = row[i];
  }
  return a;
}

// Load 32x16 B tile from PRE-TRANSPOSED f16 [N][K] storage.
// `bt` -> element (n0, k0); lane n reads 16 contiguous f16 (32 B).
__device__ __forceinline__ v16h ldBT_f16(const _Float16* bt, int ld) {
  int lane = threadIdx.x & 31;
  const _Float16* p = bt + (lane & 15) * ld + ((lane >> 4) << 4);
  v16h b;
#pragma unroll
  for (int i = 0; i < 16; ++i) b[i] = p[i];
  return b;
}

// C accumulator initialized with an N-dependent bias (bias[n0 + lane&15]).
__device__ __forceinline__ v8f initC_biasN(const float* bias_n0) {
  float v = bias_n0[threadIdx.x & 15];
  v8f c;
#pragma unroll
  for (int r = 0; r < 8; ++r) c[r] = v;
  return c;
}

__device__ __forceinline__ v8f initC_zero() {
  v8f c;
#pragma unroll
  for (int r = 0; r < 8; ++r) c[r] = 0.f;
  return c;
}

__device__ __forceinline__ void stD_f32(float* tile, int ld, v8f d, bool relu) {
  int lane = threadIdx.x & 31;
  float* p = tile + ((lane >> 4) << 3) * ld + (lane & 15);
#pragma unroll
  for (int r = 0; r < 8; ++r) p[r * ld] = relu ? fmaxf(d[r], 0.f) : d[r];
}

__device__ __forceinline__ void stD_f16(_Float16* tile, int ld, v8f d) {
  int lane = threadIdx.x & 31;
  _Float16* p = tile + ((lane >> 4) << 3) * ld + (lane & 15);
#pragma unroll
  for (int r = 0; r < 8; ++r) p[r * ld] = (_Float16)d[r];
}

// ---------------------------------------------------------------------------
// Kernel 1: fold Wq/bq into score matrices + build A48 = [Wfv; bfv; 0] (f16).
//   KW[d,j] = Wk[d,:].Wq[j,:]   kb[d] = Wk[d,:].bq
//   bw[j]   = bk.Wq[j,:]        bb    = bk.bq
// ---------------------------------------------------------------------------
__global__ void k_score_pre(const float* __restrict__ Wq, const float* __restrict__ bq,
                            const float* __restrict__ Wk, const float* __restrict__ bk,
                            const float* __restrict__ Wfv, const float* __restrict__ bfv,
                            float* __restrict__ KW, float* __restrict__ kb,
                            float* __restrict__ bw, float* __restrict__ bb,
                            _Float16* __restrict__ A48) {
  int t = threadIdx.x;
  if (t < ND * NA) {
    int d = t / NA, j = t % NA;
    float s = 0.f;
    for (int k = 0; k < KD; ++k) s += Wk[d * KD + k] * Wq[j * KD + k];
    KW[t] = s;
  } else if (t < ND * NA + ND) {
    int d = t - ND * NA;
    float s = 0.f;
    for (int k = 0; k < KD; ++k) s += Wk[d * KD + k] * bq[k];
    kb[d] = s;
  } else if (t < ND * NA + ND + NA) {
    int j = t - ND * NA - ND;
    float s = 0.f;
    for (int k = 0; k < KD; ++k) s += bk[k] * Wq[j * KD + k];
    bw[j] = s;
  } else if (t == ND * NA + ND + NA) {
    float s = 0.f;
    for (int k = 0; k < KD; ++k) s += bk[k] * bq[k];
    bb[0] = s;
  }
  for (int i = t; i < 48 * FED; i += 640) {
    int m = i / FED, f = i % FED;
    float v = (m < ND) ? Wfv[m * FED + f] : (m == ND ? bfv[f] : 0.f);
    A48[i] = (_Float16)v;
  }
}

// ---------------------------------------------------------------------------
// Kernel 2: convert + transpose weights to f16 [N][K] so WMMA B operands are
// contiguous 32-byte per-lane loads. Runs once; everything stays L2-resident.
//   W1T[h][k][v]  = W1[h][v][k]    (k<256 rows, v<192)
//   W1oT[h][k][v] = W1o[h][v][k]
//   W2T[h][v][k]  = W2[h][k][v]    (v<192 rows, k<256)
// ---------------------------------------------------------------------------
__global__ void k_cvt_weights(const float* __restrict__ W1, const float* __restrict__ W2,
                              const float* __restrict__ W1o,
                              _Float16* __restrict__ W1T, _Float16* __restrict__ W2T,
                              _Float16* __restrict__ W1oT) {
  int h = blockIdx.x;
  const float* w1  = W1  + h * VD * HID;
  const float* w1o = W1o + h * VD * HID;
  const float* w2  = W2  + h * HID * VD;
  _Float16* o1  = W1T  + h * HID * VD;
  _Float16* o1o = W1oT + h * HID * VD;
  _Float16* o2  = W2T  + h * VD * HID;
  for (int i = threadIdx.x; i < HID * VD; i += blockDim.x) {
    int k = i / VD, v = i % VD;     // o1/o1o index i = k*VD + v
    o1[i]  = (_Float16)w1[v * HID + k];
    o1o[i] = (_Float16)w1o[v * HID + k];
    int v2 = i / HID, k2 = i % HID; // o2 index i = v2*HID + k2
    o2[i]  = (_Float16)w2[k2 * VD + v2];
  }
}

// ---------------------------------------------------------------------------
// Kernel 3 (WMMA): G[h,d,k] = Wfv[d,:] @ W1[h,:128,k]
//                  c0[h,k]  = bfv @ W1[h,:128,k] + b1[h,k]
// A = A48 (48x128 f16), B = W1T[h] rows n (HID), cols f (<128). grid=32, block=256.
// Per wave: 2 n-tiles x 3 m-tiles (6 accumulators = 48 VGPRs -> no spill);
// per K-step: 3 A-loads, 2 B-loads, 6 independent WMMAs.
// ---------------------------------------------------------------------------
__global__ void k_precompute_G(const _Float16* __restrict__ A48,
                               const _Float16* __restrict__ W1T,
                               const float* __restrict__ b1,
                               float* __restrict__ G, float* __restrict__ c0) {
  int h = blockIdx.x;
  int wave = threadIdx.x >> 5;                 // 8 waves
  const _Float16* BT = W1T + h * HID * VD;     // [256][192]
  v8f acc[3][2];
#pragma unroll
  for (int mi = 0; mi < 3; ++mi)
#pragma unroll
    for (int j = 0; j < 2; ++j) acc[mi][j] = initC_zero();
#pragma unroll
  for (int ks = 0; ks < 4; ++ks) {             // K = 128
    int k0 = ks * 32;
    v16h A0 = ldA_f16(A48 +  0 * FED + k0, FED);
    v16h A1 = ldA_f16(A48 + 16 * FED + k0, FED);
    v16h A2 = ldA_f16(A48 + 32 * FED + k0, FED);
#pragma unroll
    for (int j = 0; j < 2; ++j) {
      int n0 = (wave * 2 + j) * 16;
      v16h B = ldBT_f16(BT + n0 * VD + k0, VD);
      acc[0][j] = wmma_f16(A0, B, acc[0][j]);
      acc[1][j] = wmma_f16(A1, B, acc[1][j]);
      acc[2][j] = wmma_f16(A2, B, acc[2][j]);
    }
  }
  int lane = threadIdx.x & 31;
#pragma unroll
  for (int mi = 0; mi < 3; ++mi) {
#pragma unroll
    for (int j = 0; j < 2; ++j) {
      int n = (wave * 2 + j) * 16 + (lane & 15);
      int mb = mi * 16 + ((lane >> 4) << 3);
#pragma unroll
      for (int r = 0; r < 8; ++r) {
        int m = mb + r;
        if (m < ND)       G[(h * ND + m) * HID + n] = acc[mi][j][r];
        else if (m == ND) c0[h * HID + n] = acc[mi][j][r] + b1[h * HID + n];
      }
    }
  }
}

// ---------------------------------------------------------------------------
// Kernel 4: per-batch scores -> softmax w, attn_weights output, action embeds.
// One wave per batch row. grid = 64, block = 256 (8 waves).
// ---------------------------------------------------------------------------
__global__ void k_softmax_ae(const float* __restrict__ z, const float* __restrict__ act,
                             const float* __restrict__ KW, const float* __restrict__ kb,
                             const float* __restrict__ bw, const float* __restrict__ bb,
                             const float* __restrict__ Wav, const float* __restrict__ bav,
                             float* __restrict__ w_out, _Float16* __restrict__ AE16,
                             float* __restrict__ out_attn) {
  int wave = threadIdx.x >> 5;
  int lane = threadIdx.x & 31;
  int b = blockIdx.x * 8 + wave;

  float av[NA];
#pragma unroll
  for (int j = 0; j < NA; ++j) av[j] = act[b * NA + j];

  int d = lane;  // ND == 32 == wave width
  float dotKW = 0.f, dotbw = 0.f;
#pragma unroll
  for (int j = 0; j < NA; ++j) {
    dotKW += av[j] * KW[d * NA + j];
    dotbw += av[j] * bw[j];
  }
  const float inv_sqrt_kd = 0.07216878364870323f;  // 1/sqrt(192)
  float s = (z[b * ND + d] * (dotKW + kb[d]) + (dotbw + bb[0])) * inv_sqrt_kd;

  float m = s;
  for (int off = 16; off; off >>= 1) m = fmaxf(m, __shfl_xor(m, off, 32));
  float e = __expf(s - m);
  float sum = e;
  for (int off = 16; off; off >>= 1) sum += __shfl_xor(sum, off, 32);
  float wv = e / sum;

  w_out[b * ND + d] = wv;
  for (int h = 0; h < NH; ++h) out_attn[(b * NH + h) * ND + d] = wv;

#pragma unroll
  for (int rep = 0; rep < 2; ++rep) {
    int t2 = lane + rep * 32;
    float s2 = bav[t2];
#pragma unroll
    for (int j = 0; j < NA; ++j) s2 += av[j] * Wav[j * AED + t2];
    AE16[b * AED + t2] = (_Float16)s2;
  }
}

// ---------------------------------------------------------------------------
// Kernel 5 (WMMA): base[b,h,k] = AE[b,:] @ W1[h,128:192,k] + c0[h,k]
// grid = (16 btiles of 32, 32 heads), block = 256. Per wave: 2 M x 2 N tiles.
// ---------------------------------------------------------------------------
__global__ void k_base(const _Float16* __restrict__ AE16, const _Float16* __restrict__ W1T,
                       const float* __restrict__ c0, _Float16* __restrict__ base16) {
  int b0 = blockIdx.x * 32;
  int h  = blockIdx.y;
  int wave = threadIdx.x >> 5;
  const _Float16* BT = W1T + h * HID * VD + FED;   // v-columns 128..191
  int nt0 = wave * 2;                              // 2 of 16 n-tiles
  v8f acc[2][2];
#pragma unroll
  for (int mi = 0; mi < 2; ++mi)
#pragma unroll
    for (int j = 0; j < 2; ++j) acc[mi][j] = initC_biasN(c0 + h * HID + (nt0 + j) * 16);
#pragma unroll
  for (int ks = 0; ks < 2; ++ks) {                 // K = 64
    int k0 = ks * 32;
    v16h A0 = ldA_f16(AE16 + (b0 +  0) * AED + k0, AED);
    v16h A1 = ldA_f16(AE16 + (b0 + 16) * AED + k0, AED);
#pragma unroll
    for (int j = 0; j < 2; ++j) {
      v16h B = ldBT_f16(BT + (nt0 + j) * 16 * VD + k0, VD);
      acc[0][j] = wmma_f16(A0, B, acc[0][j]);
      acc[1][j] = wmma_f16(A1, B, acc[1][j]);
    }
  }
#pragma unroll
  for (int mi = 0; mi < 2; ++mi)
#pragma unroll
    for (int j = 0; j < 2; ++j)
      stD_f16(base16 + ((b0 + mi * 16) * NH + h) * HID + (nt0 + j) * 16,
              NH * HID, acc[mi][j]);
}

// ---------------------------------------------------------------------------
// Kernel 6: hbar[b,h,k] = sum_d w[b,d] * relu(z[b,d]*G[h,d,k] + base[b,h,k])
// G[h] staged in LDS (32 KB). grid = (32 btiles of 16, 32 heads), block = 256.
// ---------------------------------------------------------------------------
__global__ void k_hbar(const float* __restrict__ z, const float* __restrict__ w,
                       const float* __restrict__ G, const _Float16* __restrict__ base16,
                       _Float16* __restrict__ hbar16) {
  __shared__ float Gl[ND * HID];
  __shared__ float zl[16 * ND];
  __shared__ float wl[16 * ND];
  int h  = blockIdx.y;
  int b0 = blockIdx.x * 16;
  int tid = threadIdx.x;  // = k index
  for (int i = tid; i < ND * HID; i += 256) Gl[i] = G[h * ND * HID + i];
  for (int i = tid; i < 16 * ND; i += 256) { zl[i] = z[b0 * ND + i]; wl[i] = w[b0 * ND + i]; }
  __syncthreads();
  for (int bi = 0; bi < 16; ++bi) {
    int b = b0 + bi;
    float bas = (float)base16[(b * NH + h) * HID + tid];
    float acc = 0.f;
#pragma unroll
    for (int d = 0; d < ND; ++d) {
      float hv = zl[bi * ND + d] * Gl[d * HID + tid] + bas;
      acc += wl[bi * ND + d] * fmaxf(hv, 0.f);
    }
    hbar16[(b * NH + h) * HID + tid] = (_Float16)acc;
  }
}

// ---------------------------------------------------------------------------
// Kernel 7 (WMMA): attn_out[b,h,v] = hbar[b,h,:] @ W2[h] + b2[h]
// grid = (16, 32), block = 256. Per wave: fixed M-tile, 3 N-tiles (1 A-load,
// 3 B-loads, 3 independent WMMAs per K-step).
// ---------------------------------------------------------------------------
__global__ void k_attn(const _Float16* __restrict__ hbar16, const _Float16* __restrict__ W2T,
                       const float* __restrict__ b2, _Float16* __restrict__ attn16) {
  int b0 = blockIdx.x * 32;
  int h  = blockIdx.y;
  int wave = threadIdx.x >> 5;
  int m0 = (wave & 1) * 16;
  int nbase = wave >> 1;                           // 0..3 of 12 n-tiles
  const _Float16* BT = W2T + h * VD * HID;         // [192][256]
  v8f acc[3];
#pragma unroll
  for (int j = 0; j < 3; ++j) acc[j] = initC_biasN(b2 + h * VD + (nbase + 4 * j) * 16);
#pragma unroll
  for (int ks = 0; ks < 8; ++ks) {                 // K = 256
    int k0 = ks * 32;
    v16h A = ldA_f16(hbar16 + ((b0 + m0) * NH + h) * HID + k0, NH * HID);
#pragma unroll
    for (int j = 0; j < 3; ++j) {
      v16h B = ldBT_f16(BT + (nbase + 4 * j) * 16 * HID + k0, HID);
      acc[j] = wmma_f16(A, B, acc[j]);
    }
  }
#pragma unroll
  for (int j = 0; j < 3; ++j)
    stD_f16(attn16 + ((b0 + m0) * NH + h) * VD + (nbase + 4 * j) * 16, NH * VD, acc[j]);
}

// ---------------------------------------------------------------------------
// Kernel 8 (WMMA): h2[b,h,k] = relu(attn_out[b,h,:] @ W1o[h] + b1o[h])
// grid = (16, 32), block = 256. Per wave: fixed M-tile, 4 N-tiles.
// ---------------------------------------------------------------------------
__global__ void k_h2(const _Float16* __restrict__ attn16, const _Float16* __restrict__ W1oT,
                     const float* __restrict__ b1o, float* __restrict__ h2) {
  int b0 = blockIdx.x * 32;
  int h  = blockIdx.y;
  int wave = threadIdx.x >> 5;
  int m0 = (wave & 1) * 16;
  int nbase = wave >> 1;                           // 0..3 of 16 n-tiles
  const _Float16* BT = W1oT + h * HID * VD;        // [256][192]
  v8f acc[4];
#pragma unroll
  for (int j = 0; j < 4; ++j) acc[j] = initC_biasN(b1o + h * HID + (nbase + 4 * j) * 16);
#pragma unroll
  for (int ks = 0; ks < 6; ++ks) {                 // K = 192
    int k0 = ks * 32;
    v16h A = ldA_f16(attn16 + ((b0 + m0) * NH + h) * VD + k0, NH * VD);
#pragma unroll
    for (int j = 0; j < 4; ++j) {
      v16h B = ldBT_f16(BT + (nbase + 4 * j) * 16 * VD + k0, VD);
      acc[j] = wmma_f16(A, B, acc[j]);
    }
  }
#pragma unroll
  for (int j = 0; j < 4; ++j)
    stD_f32(h2 + ((b0 + m0) * NH + h) * HID + (nbase + 4 * j) * 16,
            NH * HID, acc[j], /*relu=*/true);
}

// ---------------------------------------------------------------------------
// Kernel 9: effect[b,h] = h2[b,h,:].W2o[h,:] + b2o[h]  (one wave per (b,h))
// ---------------------------------------------------------------------------
__global__ void k_effect(const float* __restrict__ h2, const float* __restrict__ W2o,
                         const float* __restrict__ b2o, float* __restrict__ out_effect) {
  int p = blockIdx.x * 8 + (threadIdx.x >> 5);   // p = b*32 + h
  int lane = threadIdx.x & 31;
  int h = p & (NH - 1);
  const float* row = h2 + p * HID;
  const float* wr  = W2o + h * HID;
  float s = 0.f;
#pragma unroll
  for (int i = lane; i < HID; i += 32) s += row[i] * wr[i];
  for (int off = 16; off; off >>= 1) s += __shfl_xor(s, off, 32);
  if (lane == 0) out_effect[p] = s + b2o[h];
}

// ---------------------------------------------------------------------------
// Launch
// ---------------------------------------------------------------------------
extern "C" void kernel_launch(void* const* d_in, const int* in_sizes, int n_in,
                              void* d_out, int out_size, void* d_ws, size_t ws_size,
                              hipStream_t stream) {
  (void)in_sizes; (void)n_in; (void)out_size; (void)ws_size;
  const float* z    = (const float*)d_in[0];   // [512,32]
  const float* act  = (const float*)d_in[1];   // [512,18]
  const float* Wq   = (const float*)d_in[2];
  const float* bq   = (const float*)d_in[3];
  const float* Wk   = (const float*)d_in[4];
  const float* bk   = (const float*)d_in[5];
  const float* Wav  = (const float*)d_in[6];
  const float* bav  = (const float*)d_in[7];
  const float* Wfv  = (const float*)d_in[8];
  const float* bfv  = (const float*)d_in[9];
  const float* W1   = (const float*)d_in[10];  // [32,192,256]
  const float* b1   = (const float*)d_in[11];
  const float* W2   = (const float*)d_in[12];  // [32,256,192]
  const float* b2   = (const float*)d_in[13];
  const float* W1o  = (const float*)d_in[14];  // [32,192,256]
  const float* b1o  = (const float*)d_in[15];
  const float* W2o  = (const float*)d_in[16];  // [32,256]
  const float* b2o  = (const float*)d_in[17];  // [32]

  float* out_effect = (float*)d_out;                 // [512*32]
  float* out_attn   = (float*)d_out + NB * NH;       // [512*32*32]

  // workspace layout
  char* p = (char*)d_ws;
  float*     KW     = (float*)p;      p += 4096;
  float*     kb     = KW + ND * NA;
  float*     bw     = kb + ND;
  float*     bb     = bw + NA;
  float*     w_ws   = (float*)p;      p += NB * ND * sizeof(float);
  _Float16*  AE16   = (_Float16*)p;   p += NB * AED * sizeof(_Float16);
  float*     c0     = (float*)p;      p += NH * HID * sizeof(float);
  float*     G      = (float*)p;      p += NH * ND * HID * sizeof(float);
  _Float16*  A48    = (_Float16*)p;   p += 48 * FED * sizeof(_Float16);
  _Float16*  W1T    = (_Float16*)p;   p += (size_t)NH * HID * VD * sizeof(_Float16);
  _Float16*  W2T    = (_Float16*)p;   p += (size_t)NH * VD * HID * sizeof(_Float16);
  _Float16*  W1oT   = (_Float16*)p;   p += (size_t)NH * HID * VD * sizeof(_Float16);
  _Float16*  base16 = (_Float16*)p;   p += (size_t)NB * NH * HID * sizeof(_Float16);
  _Float16*  hbar16 = (_Float16*)p;   p += (size_t)NB * NH * HID * sizeof(_Float16);
  _Float16*  attn16 = (_Float16*)p;   p += (size_t)NB * NH * VD  * sizeof(_Float16);
  float*     h2     = (float*)p;      p += (size_t)NB * NH * HID * sizeof(float);

  k_score_pre   <<<1, 640, 0, stream>>>(Wq, bq, Wk, bk, Wfv, bfv, KW, kb, bw, bb, A48);
  k_cvt_weights <<<NH, 256, 0, stream>>>(W1, W2, W1o, W1T, W2T, W1oT);
  k_precompute_G<<<NH, 256, 0, stream>>>(A48, W1T, b1, G, c0);
  k_softmax_ae  <<<NB / 8, 256, 0, stream>>>(z, act, KW, kb, bw, bb, Wav, bav,
                                             w_ws, AE16, out_attn);
  k_base        <<<dim3(NB / 32, NH), 256, 0, stream>>>(AE16, W1T, c0, base16);
  k_hbar        <<<dim3(NB / 16, NH), 256, 0, stream>>>(z, w_ws, G, base16, hbar16);
  k_attn        <<<dim3(NB / 32, NH), 256, 0, stream>>>(hbar16, W2T, b2, attn16);
  k_h2          <<<dim3(NB / 32, NH), 256, 0, stream>>>(attn16, W1oT, b1o, h2);
  k_effect      <<<(NB * NH) / 8, 256, 0, stream>>>(h2, W2o, b2o, out_effect);
}
